// LineVectorizer_24893630447786
// MI455X (gfx1250) — compile-verified, matrix operand order checked by reading, not run on hardware
//
#include <hip/hip_runtime.h>
#include <hip/hip_bf16.h>

#define BATCH   4
#define NLINE   5000
#define HH      256
#define WW      256
#define HWPIX   65536
#define CIN     256
#define DLOI    128
#define NPTS0   32
#define NPTS1   8
#define DFC     1024
#define NOUT    2500

typedef __attribute__((ext_vector_type(16))) _Float16 v16h;
typedef __attribute__((ext_vector_type(8)))  float    v8f;

union Frag { v16h h; unsigned u[8]; };

// CDNA5 async global->LDS copy path (ASYNCcnt-tracked), with safe fallback.
// Probe round 2 showed the builtin takes v4i pointers: arg0 = global src
// (addrspace(1), printed as "__device__"), arg1 = LDS dst (addrspace(3)).
#if defined(__gfx1250__) && __has_builtin(__builtin_amdgcn_global_load_async_to_lds_b128) && __has_builtin(__builtin_amdgcn_s_wait_asynccnt)
#define USE_ASYNC_LDS 1
typedef int v4i_ __attribute__((ext_vector_type(4)));
typedef __attribute__((address_space(1))) v4i_* gv4p;
typedef __attribute__((address_space(3))) v4i_* lv4p;
#else
#define USE_ASYNC_LDS 0
#endif

// ---------------------------------------------------------------------------
// Generic f16 GEMM with f32 accumulate: C[M,N] = relu?(A[M,K] @ Bt[N,K]^T + bias)
// 256 threads = 8 waves; block tile 128(M) x 64(N); K staged in 32-wide steps
// through double-buffered LDS; next tile's async copy overlaps current WMMAs.
// ---------------------------------------------------------------------------
__global__ __launch_bounds__(256) void gemm_wmma_f16(
    const _Float16* __restrict__ A,   // [M,K] row-major
    const _Float16* __restrict__ Bt,  // [N,K] row-major (B transposed)
    const float*    __restrict__ bias,// [N]
    _Float16*       __restrict__ C,   // [M,N] row-major
    int M, int N, int K, int relu)
{
  __shared__ _Float16 sA[2][128 * 32];
  __shared__ _Float16 sB[2][64 * 32];

  const int t    = threadIdx.x;
  const int wave = t >> 5;
  const int lane = t & 31;
  const int l16  = lane & 15;
  const int kh   = lane >> 4;          // K-half selector per WMMA A/B layout
  const int m0   = blockIdx.x * 128;
  const int n0   = blockIdx.y * 64;

  // per-thread 16-byte staging chunks: A = 128 rows x 4 chunks (2/thread),
  // B = 64 rows x 4 chunks (1/thread)
  const int srow = t >> 2;             // 0..63
  const int kc   = (t & 3) * 8;        // halves within row
  int gm0 = m0 + srow;      if (gm0 >= M) gm0 = M - 1;
  int gm1 = m0 + 64 + srow; if (gm1 >= M) gm1 = M - 1;

  auto stage = [&](int buf, int kb) {
#if USE_ASYNC_LDS
    __builtin_amdgcn_global_load_async_to_lds_b128(
        (gv4p)(void*)&A[(size_t)gm0 * K + kb + kc],
        (lv4p)(void*)&sA[buf][srow * 32 + kc], 0, 0);
    __builtin_amdgcn_global_load_async_to_lds_b128(
        (gv4p)(void*)&A[(size_t)gm1 * K + kb + kc],
        (lv4p)(void*)&sA[buf][(64 + srow) * 32 + kc], 0, 0);
    __builtin_amdgcn_global_load_async_to_lds_b128(
        (gv4p)(void*)&Bt[(size_t)(n0 + srow) * K + kb + kc],
        (lv4p)(void*)&sB[buf][srow * 32 + kc], 0, 0);
#else
    *(uint4*)&sA[buf][srow * 32 + kc] =
        *(const uint4*)&A[(size_t)gm0 * K + kb + kc];
    *(uint4*)&sA[buf][(64 + srow) * 32 + kc] =
        *(const uint4*)&A[(size_t)gm1 * K + kb + kc];
    *(uint4*)&sB[buf][srow * 32 + kc] =
        *(const uint4*)&Bt[(size_t)(n0 + srow) * K + kb + kc];
#endif
  };

  v8f acc[4] = {v8f{}, v8f{}, v8f{}, v8f{}};

  const int nk = K / 32;
  stage(0, 0);
  for (int it = 0; it < nk; ++it) {
    const int cur = it & 1;
    const bool has_next = (it + 1) < nk;
    if (has_next) stage(1 - cur, (it + 1) * 32);   // overlaps with WMMAs below
#if USE_ASYNC_LDS
    // async loads complete in order per wave: leaving the 3 newest outstanding
    // guarantees the current buffer (3 older ops) has landed in LDS.
    if (has_next) __builtin_amdgcn_s_wait_asynccnt(3);
    else          __builtin_amdgcn_s_wait_asynccnt(0);
#endif
    __syncthreads();

    // A fragment (16x32 f16): lane = M + 16*khalf; VGPR j holds K pairs
    Frag a;
    {
      const _Float16* ar = &sA[cur][(wave * 16 + l16) * 32];
#pragma unroll
      for (int j = 0; j < 4; ++j) {
        a.u[j]     = *(const unsigned*)&ar[kh * 8 + 2 * j];
        a.u[4 + j] = *(const unsigned*)&ar[16 + kh * 8 + 2 * j];
      }
    }
#pragma unroll
    for (int ts = 0; ts < 4; ++ts) {
      Frag b;
      const _Float16* br = &sB[cur][(ts * 16 + l16) * 32];
#pragma unroll
      for (int j = 0; j < 4; ++j) {
        b.u[j]     = *(const unsigned*)&br[kh * 8 + 2 * j];
        b.u[4 + j] = *(const unsigned*)&br[16 + kh * 8 + 2 * j];
      }
      acc[ts] = __builtin_amdgcn_wmma_f32_16x16x32_f16(
          false, a.h, false, b.h, (short)0, acc[ts], false, false);
    }
    __syncthreads();   // readers done before this buffer is restaged
  }

  // Epilogue: D layout — VGPR r: lanes 0-15 -> M=r, lanes 16-31 -> M=8+r; N=lane%16
  const int mbase = m0 + wave * 16 + kh * 8;
#pragma unroll
  for (int ts = 0; ts < 4; ++ts) {
    int n   = n0 + ts * 16 + l16;
    float bv = bias ? bias[n] : 0.0f;
#pragma unroll
    for (int r = 0; r < 8; ++r) {
      int m = mbase + r;
      if (m < M) {
        float v = acc[ts][r] + bv;
        if (relu && v < 0.0f) v = 0.0f;
        C[(size_t)m * N + n] = (_Float16)v;
      }
    }
  }
}

// ---------------------------------------------------------------------------
// feature [B,C,H,W] f32  ->  featT [B, H*W, C] f16  (LDS-tiled transpose)
// ---------------------------------------------------------------------------
__global__ __launch_bounds__(256) void transpose_feat(
    const float* __restrict__ f, _Float16* __restrict__ ft)
{
  __shared__ float tile[32][33];
  int b  = blockIdx.z;
  int p0 = blockIdx.x * 32;
  int c0 = blockIdx.y * 32;
  int tx = threadIdx.x & 31;
  int ty = threadIdx.x >> 5;           // 0..7
  const float* fb = f + (size_t)b * CIN * HWPIX;
#pragma unroll
  for (int i = 0; i < 4; ++i) {
    int c = c0 + ty + 8 * i;
    tile[ty + 8 * i][tx] = fb[(size_t)c * HWPIX + p0 + tx];
  }
  __syncthreads();
  _Float16* ftb = ft + (size_t)b * HWPIX * CIN;
#pragma unroll
  for (int i = 0; i < 4; ++i) {
    int p = p0 + ty + 8 * i;
    ftb[(size_t)p * CIN + c0 + tx] = (_Float16)tile[tx][ty + 8 * i];
  }
}

// ---------------------------------------------------------------------------
// f32 -> f16 copy (fc1_w is already [N=DLOI][K=CIN] = Bt layout)
// ---------------------------------------------------------------------------
__global__ void cvt_f32_f16(const float* __restrict__ in,
                            _Float16* __restrict__ out, int n)
{
  int i = blockIdx.x * 256 + threadIdx.x;
  if (i < n) out[i] = (_Float16)in[i];
}

// ---------------------------------------------------------------------------
// w [K,N] f32 -> wt [N,K] f16 (LDS-tiled transpose)
// ---------------------------------------------------------------------------
__global__ __launch_bounds__(256) void transpose_w(
    const float* __restrict__ w, _Float16* __restrict__ wt, int K, int N)
{
  __shared__ float tile[32][33];
  int k0 = blockIdx.x * 32;
  int n0 = blockIdx.y * 32;
  int tx = threadIdx.x & 31;
  int ty = threadIdx.x >> 5;
#pragma unroll
  for (int i = 0; i < 4; ++i)
    tile[ty + 8 * i][tx] = w[(size_t)(k0 + ty + 8 * i) * N + n0 + tx];
  __syncthreads();
#pragma unroll
  for (int i = 0; i < 4; ++i)
    wt[(size_t)(n0 + ty + 8 * i) * K + k0 + tx] = (_Float16)tile[tx][ty + 8 * i];
}

// ---------------------------------------------------------------------------
// Bilinear sampling along lines + maxpool(4) -> feat[line][o*8+p1] (f16)
// One block (128 threads = channels) per line; point geometry cached in LDS.
// x_t layout [b*HWPIX + pix][128] makes every tap a coalesced 256B read.
// ---------------------------------------------------------------------------
__global__ __launch_bounds__(128) void sample_pool(
    const _Float16* __restrict__ xt,     // [B*HWPIX, DLOI] f16
    const float*    __restrict__ lines,  // [B, L, 2, 2]
    _Float16*       __restrict__ feat)   // [B*L, DLOI*NPTS1]
{
  int gl = blockIdx.x;                   // 0 .. B*L-1
  int b  = gl / NLINE;
  int o  = threadIdx.x;                  // channel

  __shared__ int   sIdx[NPTS0][4];
  __shared__ float sW[NPTS0][4];

  const float* ln = lines + (size_t)gl * 4;
  if (threadIdx.x < NPTS0) {
    int i = threadIdx.x;
    float ax = ln[0], ay = ln[1], bx = ln[2], by = ln[3];
    float lam = (float)i / (float)(NPTS0 - 1);
    float px = ax * lam + bx * (1.0f - lam) - 0.5f;
    float py = ay * lam + by * (1.0f - lam) - 0.5f;
    float px0 = fminf(fmaxf(floorf(px), 0.0f), (float)(HH - 1));
    float py0 = fminf(fmaxf(floorf(py), 0.0f), (float)(WW - 1));
    float px1 = fminf(px0 + 1.0f, (float)(HH - 1));
    float py1 = fminf(py0 + 1.0f, (float)(WW - 1));
    int ix0 = (int)px0, iy0 = (int)py0, ix1 = (int)px1, iy1 = (int)py1;
    sIdx[i][0] = ix0 * WW + iy0;
    sIdx[i][1] = ix1 * WW + iy0;
    sIdx[i][2] = ix0 * WW + iy1;
    sIdx[i][3] = ix1 * WW + iy1;
    sW[i][0] = (px1 - px) * (py1 - py);
    sW[i][1] = (px - px0) * (py1 - py);
    sW[i][2] = (px1 - px) * (py - py0);
    sW[i][3] = (px - px0) * (py - py0);
  }
  __syncthreads();

  const _Float16* xb = xt + (size_t)b * HWPIX * DLOI;
  _Float16* fo = feat + (size_t)gl * (DLOI * NPTS1);
#pragma unroll
  for (int p1 = 0; p1 < NPTS1; ++p1) {
    float mx = -3.0e38f;
#pragma unroll
    for (int s = 0; s < 4; ++s) {
      int pt = p1 * 4 + s;
      float v = 0.0f;
#pragma unroll
      for (int q = 0; q < 4; ++q)
        v += sW[pt][q] * (float)xb[(size_t)sIdx[pt][q] * DLOI + o];
      mx = fmaxf(mx, v);
    }
    fo[o * NPTS1 + p1] = (_Float16)mx;   // torch-style channel-major packing
  }
}

// ---------------------------------------------------------------------------
// logits = h2 @ w3 + b3 -> softmax -> s_out; key = keep ? argmax : -1
// One wave32 per line, shuffle reduction.
// ---------------------------------------------------------------------------
__global__ __launch_bounds__(256) void logits_softmax(
    const _Float16* __restrict__ h2,   // [B*L, DFC]
    const float*    __restrict__ w3,   // [DFC, 4]
    const float*    __restrict__ b3,   // [4]
    float*          __restrict__ s_out,// [B*L, 4]
    int*            __restrict__ keys) // [B*L]
{
  int wave = threadIdx.x >> 5;
  int lane = threadIdx.x & 31;
  int l = blockIdx.x * 8 + wave;
  if (l >= BATCH * NLINE) return;

  const _Float16* hr = h2 + (size_t)l * DFC;
  float d0 = 0.f, d1 = 0.f, d2 = 0.f, d3 = 0.f;
  for (int k = lane; k < DFC; k += 32) {
    float h = (float)hr[k];
    const float* wr = w3 + (size_t)k * 4;
    d0 += h * wr[0]; d1 += h * wr[1]; d2 += h * wr[2]; d3 += h * wr[3];
  }
#pragma unroll
  for (int off = 16; off > 0; off >>= 1) {
    d0 += __shfl_down(d0, off, 32);
    d1 += __shfl_down(d1, off, 32);
    d2 += __shfl_down(d2, off, 32);
    d3 += __shfl_down(d3, off, 32);
  }
  if (lane == 0) {
    float lg[4] = {d0 + b3[0], d1 + b3[1], d2 + b3[2], d3 + b3[3]};
    float mx = fmaxf(fmaxf(lg[0], lg[1]), fmaxf(lg[2], lg[3]));
    float e[4], sum = 0.f;
#pragma unroll
    for (int c = 0; c < 4; ++c) { e[c] = __expf(lg[c] - mx); sum += e[c]; }
    float inv = 1.0f / sum;
    float sv[4]; int best = 0; float bv = -1.0f;
#pragma unroll
    for (int c = 0; c < 4; ++c) {
      sv[c] = e[c] * inv;
      s_out[(size_t)l * 4 + c] = sv[c];
      if (sv[c] > bv) { bv = sv[c]; best = c; }   // first max wins ties
    }
    bool keep = ((sv[1] > 0.25f) || (sv[2] > 0.25f) || (sv[3] > 0.25f)) &&
                (sv[0] < 0.25f);
    keys[l] = keep ? best : -1;
  }
}

// ---------------------------------------------------------------------------
// Stable descending sort by key in {3,2,1,0,-1} == stable counting sort.
// One block per batch: per-thread contiguous chunks (stability) + LDS scan.
// ---------------------------------------------------------------------------
__global__ __launch_bounds__(1024) void select_kernel(
    const int*   __restrict__ keys,      // [B*L]
    const float* __restrict__ lines,     // [B,L,4]
    const float* __restrict__ s,         // [B*L,4]
    float*       __restrict__ lines_out, // [B,NOUT,4]
    float*       __restrict__ score_out) // [B,NOUT,4]
{
  int b = blockIdx.x;
  int t = threadIdx.x;
  __shared__ int order[NLINE];
  __shared__ int scan[1024];
  __shared__ int bucket_total[5];
  __shared__ int base[5];

  const int* kb = keys + b * NLINE;
  int lo = t * 5;
  int hi = lo + 5; if (hi > NLINE) hi = NLINE;
  if (lo > NLINE) lo = NLINE;

  int cnt_local[5] = {0, 0, 0, 0, 0};
  for (int i = lo; i < hi; ++i) {
    int key = kb[i];
    int bkt = (key < 0) ? 4 : (3 - key);   // desc order: 3,2,1,0,-1
    cnt_local[bkt]++;
  }

  int excl[5];
  for (int bkt = 0; bkt < 5; ++bkt) {
    scan[t] = cnt_local[bkt];
    __syncthreads();
    for (int off = 1; off < 1024; off <<= 1) {
      int v = (t >= off) ? scan[t - off] : 0;
      __syncthreads();
      scan[t] += v;
      __syncthreads();
    }
    excl[bkt] = scan[t] - cnt_local[bkt];
    if (t == 1023) bucket_total[bkt] = scan[1023];
    __syncthreads();
  }
  if (t == 0) {
    int acc = 0;
    for (int bkt = 0; bkt < 5; ++bkt) { base[bkt] = acc; acc += bucket_total[bkt]; }
  }
  __syncthreads();

  int pos[5];
#pragma unroll
  for (int bkt = 0; bkt < 5; ++bkt) pos[bkt] = base[bkt] + excl[bkt];
  for (int i = lo; i < hi; ++i) {
    int key = kb[i];
    int bkt = (key < 0) ? 4 : (3 - key);
    order[pos[bkt]++] = i;
  }
  __syncthreads();

  int cnt = NLINE - bucket_total[4];
  int denom = cnt > 0 ? cnt : 1;
  for (int i = t; i < NOUT; i += 1024) {
    int idx = order[i % denom];
    const float* lp = lines + ((size_t)b * NLINE + idx) * 4;
    const float* sp = s     + ((size_t)b * NLINE + idx) * 4;
#pragma unroll
    for (int c = 0; c < 4; ++c) {
      lines_out[((size_t)b * NOUT + i) * 4 + c] = (cnt > 0) ? lp[c] : 0.0f;
      score_out[((size_t)b * NOUT + i) * 4 + c] = (cnt > 0) ? sp[c] : 0.0f;
    }
  }
}

// ---------------------------------------------------------------------------
extern "C" void kernel_launch(void* const* d_in, const int* in_sizes, int n_in,
                              void* d_out, int out_size, void* d_ws, size_t ws_size,
                              hipStream_t stream) {
  const float* feature = (const float*)d_in[0];
  const float* lines   = (const float*)d_in[1];
  const float* fc1_w   = (const float*)d_in[2];
  const float* fc1_b   = (const float*)d_in[3];
  const float* w1      = (const float*)d_in[4];
  const float* b1      = (const float*)d_in[5];
  const float* w2      = (const float*)d_in[6];
  const float* b2      = (const float*)d_in[7];
  const float* w3      = (const float*)d_in[8];
  const float* b3      = (const float*)d_in[9];

  float* out       = (float*)d_out;
  float* lines_out = out;                    // [4,2500,2,2]
  float* score_out = out + 40000;            // [4,2500,4]
  float* s_out     = out + 80000;            // [4,5000,4]

  char* ws = (char*)d_ws;
  // region A: featT (128 MB), later aliased by feat/h1/h2 (3x40 MB)
  _Float16* featT  = (_Float16*)(ws + 0);
  _Float16* feat   = (_Float16*)(ws + 0);
  _Float16* h1     = (_Float16*)(ws + 41943040ull);
  _Float16* h2     = (_Float16*)(ws + 83886080ull);
  // region B: x_t (64 MB)
  _Float16* xt     = (_Float16*)(ws + 134217728ull);
  // region C: f16 weights + keys
  _Float16* fc1w16 = (_Float16*)(ws + 201326592ull);
  _Float16* w1t    = (_Float16*)(ws + 201392128ull);
  _Float16* w2t    = (_Float16*)(ws + 203489280ull);
  int*      keys   = (int*)     (ws + 205586432ull);

  // 1) feature [B,C,HW] f32 -> featT [B,HW,C] f16
  transpose_feat<<<dim3(HWPIX / 32, CIN / 32, BATCH), 256, 0, stream>>>(feature, featT);
  // 2) weights to f16 (Bt layouts)
  cvt_f32_f16<<<(DLOI * CIN + 255) / 256, 256, 0, stream>>>(fc1_w, fc1w16, DLOI * CIN);
  transpose_w<<<dim3(DFC / 32, DFC / 32), 256, 0, stream>>>(w1, w1t, DFC, DFC);
  transpose_w<<<dim3(DFC / 32, DFC / 32), 256, 0, stream>>>(w2, w2t, DFC, DFC);
  // 3) fc1 GEMM: x_t[B*HW, 128] = featT @ fc1_w^T + fc1_b
  gemm_wmma_f16<<<dim3(BATCH * HWPIX / 128, DLOI / 64), 256, 0, stream>>>(
      featT, fc1w16, fc1_b, xt, BATCH * HWPIX, DLOI, CIN, 0);
  // 4) bilinear sample + maxpool -> feat [20000, 1024]
  sample_pool<<<BATCH * NLINE, 128, 0, stream>>>(xt, lines, feat);
  // 5) MLP GEMMs with relu
  gemm_wmma_f16<<<dim3((BATCH * NLINE + 127) / 128, DFC / 64), 256, 0, stream>>>(
      feat, w1t, b1, h1, BATCH * NLINE, DFC, DFC, 1);
  gemm_wmma_f16<<<dim3((BATCH * NLINE + 127) / 128, DFC / 64), 256, 0, stream>>>(
      h1, w2t, b2, h2, BATCH * NLINE, DFC, DFC, 1);
  // 6) logits + softmax + keep/argmax keys
  logits_softmax<<<(BATCH * NLINE) / 8, 256, 0, stream>>>(h2, w3, b3, s_out, keys);
  // 7) stable counting-sort select per batch
  select_kernel<<<BATCH, 1024, 0, stream>>>(keys, lines, s_out, lines_out, score_out);
}
